// QMOILongformerAttention_43327630082343
// MI455X (gfx1250) — compile-verified, matrix operand order checked by reading, not run on hardware
//
#include <hip/hip_runtime.h>
#include <hip/hip_bf16.h>

#define S_LEN 2048
#define HIDN  1024
#define NH    16
#define HD    64
#define WIN   512
#define NWIN  4
#define LDC   516   // padded f32 row stride for attention score strip (16B-aligned rows)
#define GST   40    // padded f16 row stride for GEMM LDS tiles (80B: 16B-aligned, conflict-free)

typedef _Float16 v16h __attribute__((ext_vector_type(16)));
typedef float    v8f  __attribute__((ext_vector_type(8)));

// Load a 16-bit A/B WMMA fragment from a row-major (K-contiguous) row pointer.
// CDNA5 layout: lanes 0-15 hold K[kb..kb+7] + K[kb+16..kb+23] with kb=0,
// lanes 16-31 the same with kb=8. Caller passes kbase = k0 + (lane>=16 ? 8 : 0).
__device__ __forceinline__ v16h load_frag(const _Float16* __restrict__ row, int kbase) {
    union { int4 q[2]; v16h h; } u;
    u.q[0] = *reinterpret_cast<const int4*>(row + kbase);
    u.q[1] = *reinterpret_cast<const int4*>(row + kbase + 16);
    return u.h;
}

__device__ __forceinline__ v8f wmma32(v16h a, v16h b, v8f c) {
    return __builtin_amdgcn_wmma_f32_16x16x32_f16(false, a, false, b, (short)0, c, false, false);
}

// Async-copy 32B (two per-lane b128 chunks) global -> LDS. ISA adds INST_OFFSET
// to BOTH the LDS and global addresses, so offset:16 covers the second chunk.
__device__ __forceinline__ void async_copy_32B(unsigned ldsOff, const _Float16* gaddr) {
    asm volatile(
        "global_load_async_to_lds_b128 %0, %1, off\n\t"
        "global_load_async_to_lds_b128 %0, %1, off offset:16"
        :: "v"(ldsOff), "v"(gaddr)
        : "memory");
}

__global__ void cvt_f32_f16_kernel(const float* __restrict__ in, _Float16* __restrict__ out, int n) {
    int i = blockIdx.x * blockDim.x + threadIdx.x;
    const int stride = gridDim.x * blockDim.x;
    for (; i < n; i += stride) {
        __builtin_prefetch(in + i + 4 * stride, 0, 0);   // global_prefetch_b8
        out[i] = (_Float16)in[i];
    }
}

// C = A(MxK, f16 row-major) x Bt(NxK, f16 row-major)^T + bias.
// A/B 128x32 tiles staged in LDS via async copies, double-buffered on ASYNCcnt.
// MODE 0: store f16 at [h][m][d] with n = h*64+d   (Q, K)
// MODE 1: store f16 transposed at [n][m]           (V^T: [h][d][s])
// MODE 2: store f32 row-major [m][n]               (final projection)
template <int MODE>
__global__ void gemm_wmma_kernel(const _Float16* __restrict__ A,
                                 const _Float16* __restrict__ Bt,
                                 const float* __restrict__ bias,
                                 void* __restrict__ outv,
                                 int M, int N, int K) {
    __shared__ _Float16 sA[2][128 * GST];
    __shared__ _Float16 sB[2][128 * GST];

    const int tid  = threadIdx.x;
    const int w    = tid >> 5;
    const int lane = tid & 31;
    const int lm   = lane & 15;
    const int hi   = lane >> 4;
    const int koff = hi * 8;
    const int mBlk = blockIdx.y * 128;
    const int nBlk = blockIdx.x * 128;
    const int mbase = mBlk + (w & 3) * 32;
    const int nbase = nBlk + (w >> 2) * 64;

    // tile-copy assignment: 256 threads x 32B = one 8KB (128x32 f16) tile
    const int crow = tid >> 1;
    const int ccol = (tid & 1) * 16;
    const unsigned laBase = (unsigned)(uintptr_t)(&sA[0][0]) + (unsigned)(crow * GST + ccol) * 2u;
    const unsigned lbBase = (unsigned)(uintptr_t)(&sB[0][0]) + (unsigned)(crow * GST + ccol) * 2u;
    const unsigned bufStep = (unsigned)(128 * GST) * 2u;
    const _Float16* gaRow = A  + (size_t)(mBlk + crow) * K + ccol;
    const _Float16* gbRow = Bt + (size_t)(nBlk + crow) * K + ccol;

    v8f zero = {};
    v8f acc[2][4];
#pragma unroll
    for (int mt = 0; mt < 2; ++mt)
#pragma unroll
        for (int nt = 0; nt < 4; ++nt) acc[mt][nt] = zero;

    const int nsteps = K / 32;
    // stage tile 0 (4 async instrs per wave -> ASYNCcnt = 4)
    async_copy_32B(laBase, gaRow);
    async_copy_32B(lbBase, gbRow);

    for (int j = 0; j < nsteps; ++j) {
        const int cur = j & 1;
        if (j + 1 < nsteps) {
            // stage next tile into the other buffer, then drain only the current tile
            async_copy_32B(laBase + (cur ^ 1) * bufStep, gaRow + (j + 1) * 32);
            async_copy_32B(lbBase + (cur ^ 1) * bufStep, gbRow + (j + 1) * 32);
            asm volatile("s_wait_asynccnt 4" ::: "memory");  // current 4 done, next 4 in flight
        } else {
            asm volatile("s_wait_asynccnt 0" ::: "memory");
        }
        __syncthreads();   // all waves' shares of the current tile are in LDS

        const _Float16* a0 = &sA[cur][((w & 3) * 32 + lm) * GST];
        const _Float16* a1 = &sA[cur][((w & 3) * 32 + 16 + lm) * GST];
        v16h fa0 = load_frag(a0, koff);
        v16h fa1 = load_frag(a1, koff);
#pragma unroll
        for (int nt = 0; nt < 4; ++nt) {
            const _Float16* b = &sB[cur][((w >> 2) * 64 + nt * 16 + lm) * GST];
            v16h fb = load_frag(b, koff);
            acc[0][nt] = wmma32(fa0, fb, acc[0][nt]);
            acc[1][nt] = wmma32(fa1, fb, acc[1][nt]);
        }
        __syncthreads();   // everyone done reading buf[cur] before it is overwritten
    }

#pragma unroll
    for (int mt = 0; mt < 2; ++mt) {
#pragma unroll
        for (int nt = 0; nt < 4; ++nt) {
            const int ncol  = nbase + nt * 16 + lm;     // C layout: lane owns column
            const float bv  = bias[ncol];
            const int mrow0 = mbase + mt * 16 + hi * 8; // rows mrow0..mrow0+7 across acc regs
            if (MODE == 0) {
                _Float16* out = (_Float16*)outv;
                const int hh = ncol >> 6, dd = ncol & 63;
#pragma unroll
                for (int i = 0; i < 8; ++i)
                    out[((size_t)hh * M + (mrow0 + i)) * 64 + dd] = (_Float16)(acc[mt][nt][i] + bv);
            } else if (MODE == 1) {
                _Float16* out = (_Float16*)outv;
                union { _Float16 h[8]; int4 q; } u;
#pragma unroll
                for (int i = 0; i < 8; ++i) u.h[i] = (_Float16)(acc[mt][nt][i] + bv);
                *reinterpret_cast<int4*>(out + (size_t)ncol * M + mrow0) = u.q;  // contiguous b128
            } else {
                float* out = (float*)outv;
#pragma unroll
                for (int i = 0; i < 8; ++i)
                    out[(size_t)(mrow0 + i) * N + ncol] = acc[mt][nt][i] + bv;
            }
        }
    }
}

// Per (h,d): total V sum over all s, and per-window sums. vT is [H*D][S] f16.
__global__ void reduce_v_kernel(const _Float16* __restrict__ vT,
                                float* __restrict__ vtot, float* __restrict__ vwin) {
    const int hd = blockIdx.x;   // 0..1023
    const int t  = threadIdx.x;  // 256
    const _Float16* row = vT + (size_t)hd * S_LEN;
    float s = 0.0f;
#pragma unroll
    for (int i = 0; i < 8; ++i) s += (float)row[t * 8 + i];
    __shared__ float part[256];
    __shared__ float wsum[4];
    part[t] = s;
    __syncthreads();
    if (t < 4) {
        float x = 0.0f;
        for (int i = 0; i < 64; ++i) x += part[t * 64 + i];
        wsum[t] = x;
        vwin[(t * NH + (hd >> 6)) * 64 + (hd & 63)] = x;   // [nW][H][D]
    }
    __syncthreads();
    if (t == 0) vtot[hd] = wsum[0] + wsum[1] + wsum[2] + wsum[3];
}

// One workgroup = 64 query rows of (window nwi, head h). 8 waves.
// Phase 1: S = scale * Q Kt (doubled on global-global in-window pairs) -> LDS f32 64x512
// Phase 2: modified softmax stats (implicit-zero columns, extra global columns)
// Phase 3: O = P V via WMMA + corrections; store f16 attention output [s][hid]
__global__ void attn_kernel(const _Float16* __restrict__ qh,
                            const _Float16* __restrict__ kh,
                            const _Float16* __restrict__ vT,
                            const float* __restrict__ vtot,
                            const float* __restrict__ vwin,
                            _Float16* __restrict__ aout) {
    extern __shared__ float smem[];
    float* sc     = smem;              // [64][LDC] raw scores, then exp(s-m)
    float* rowZ   = smem + 64 * LDC;   // [64]
    float* rowEM  = rowZ + 64;         // [64] exp(-m)
    float* extraE = rowEM + 64;        // [2][64] extra-column weights for rows 0 and 32

    const int blk  = blockIdx.x;   // 0..7   (64-row block within window)
    const int h    = blockIdx.y;   // 0..15
    const int nwi  = blockIdx.z;   // 0..3
    const int w    = threadIdx.x >> 5;
    const int lane = threadIdx.x & 31;
    const int lm   = lane & 15;
    const int hi   = lane >> 4;
    const int koff = hi * 8;
    const float scale = 0.125f;            // D^-0.5, D=64
    const int s0 = nwi * WIN + blk * 64;   // absolute index of first row in block

    // ---------------- phase 1: score strip ----------------
    {
        const int rAbs = s0 + (w & 3) * 16 + lm;
        const _Float16* qrow = qh + ((size_t)h * S_LEN + rAbs) * HD;
        v16h aq0 = load_frag(qrow, koff);        // K 0..31
        v16h aq1 = load_frag(qrow, 32 + koff);   // K 32..63
        const int rA0 = s0 + (w & 3) * 16 + hi * 8;
        for (int tt = 0; tt < 16; ++tt) {
            const int colL   = (w >> 2) * 256 + tt * 16 + lm;
            const int colAbs = nwi * WIN + colL;
            const _Float16* krow = kh + ((size_t)h * S_LEN + colAbs) * HD;
            v16h b0 = load_frag(krow, koff);
            v16h b1 = load_frag(krow, 32 + koff);
            v8f accv = {};
            accv = wmma32(aq0, b0, accv);
            accv = wmma32(aq1, b1, accv);
            const bool colG = ((colAbs & 31) == 0);
#pragma unroll
            for (int i = 0; i < 8; ++i) {
                float s = accv[i] * scale;
                if (colG && (((rA0 + i) & 31) == 0)) s *= 2.0f;  // global pair in-window: local+global
                sc[((w & 3) * 16 + hi * 8 + i) * LDC + colL] = s;
            }
        }
    }
    __syncthreads();

    // ---------------- phase 2: softmax statistics ----------------
    for (int rr = 0; rr < 8; ++rr) {
        const int lr = w * 8 + rr;
        float* rowp = sc + lr * LDC;
        float mx = 0.0f;   // implicit exp(0) columns participate in the max
        for (int c = lane; c < WIN; c += 32) mx = fmaxf(mx, rowp[c]);
        const bool isG = ((lr & 31) == 0);   // global rows land on local rows 0 and 32
        float garr[2]; bool gok[2] = {false, false};
        if (isG) {
            const int iAbs = s0 + lr;
            const _Float16* qrow = qh + ((size_t)h * S_LEN + iAbs) * HD;
#pragma unroll
            for (int rep = 0; rep < 2; ++rep) {
                const int t = lane + rep * 32;      // global token id 0..63, col = 32*t
                if ((t >> 4) != nwi) {              // only columns outside our window
                    const _Float16* krow = kh + ((size_t)h * S_LEN + t * 32) * HD;
                    float dot = 0.0f;
                    for (int d = 0; d < HD; ++d) dot += (float)qrow[d] * (float)krow[d];
                    garr[rep] = dot * scale;
                    gok[rep] = true;
                    mx = fmaxf(mx, garr[rep]);
                }
            }
        }
#pragma unroll
        for (int off = 16; off > 0; off >>= 1) mx = fmaxf(mx, __shfl_xor(mx, off));
        const float em = __expf(-mx);
        float zs = 0.0f;
        for (int c = lane; c < WIN; c += 32) {
            const float p = __expf(rowp[c] - mx);
            rowp[c] = p;
            zs += p;
        }
        if (isG) {
#pragma unroll
            for (int rep = 0; rep < 2; ++rep) {
                const int t = lane + rep * 32;
                float e = 0.0f;
                if (gok[rep]) e = __expf(garr[rep] - mx) - em;  // replaces one exp(0) slot
                extraE[((lr >> 5) & 1) * 64 + t] = e;
                zs += e;
            }
        }
#pragma unroll
        for (int off = 16; off > 0; off >>= 1) zs += __shfl_xor(zs, off);
        if (lane == 0) {
            rowZ[lr]  = zs + (float)(S_LEN - WIN) * em;  // masked columns contribute exp(0-m)
            rowEM[lr] = em;
        }
    }
    __syncthreads();

    // ---------------- phase 3: O = P @ V + corrections ----------------
    const int r  = w & 3;
    const int c0 = (w >> 2) * 32;
    v8f o0 = {}, o1 = {};
    {
        const float* prow = sc + (r * 16 + lm) * LDC;
        const _Float16* vrow0 = vT + ((size_t)(h * 64 + c0 + lm)) * S_LEN + nwi * WIN;
        const _Float16* vrow1 = vT + ((size_t)(h * 64 + c0 + 16 + lm)) * S_LEN + nwi * WIN;
        for (int kk = 0; kk < 16; ++kk) {
            const int jb = kk * 32 + koff;
            v16h pa;
#pragma unroll
            for (int i = 0; i < 8; ++i) {
                pa[i]     = (_Float16)prow[jb + i];
                pa[i + 8] = (_Float16)prow[jb + 16 + i];
            }
            v16h vb0 = load_frag(vrow0, kk * 32 + koff);
            v16h vb1 = load_frag(vrow1, kk * 32 + koff);
            o0 = wmma32(pa, vb0, o0);
            o1 = wmma32(pa, vb1, o1);
        }
    }
#pragma unroll
    for (int dt = 0; dt < 2; ++dt) {
        const int d = c0 + dt * 16 + lm;
        v8f o = dt ? o1 : o0;
        const float vt = vtot[h * 64 + d];
        const float vw = vwin[(nwi * NH + h) * 64 + d];
        const _Float16* vcol = vT + ((size_t)(h * 64 + d)) * S_LEN;
#pragma unroll
        for (int i = 0; i < 8; ++i) {
            const int lr = r * 16 + hi * 8 + i;
            float val = o[i] + rowEM[lr] * (vt - vw);   // implicit-1 columns outside window
            if ((lr & 31) == 0) {                        // global row: extra global columns
                const float* ep = extraE + ((lr >> 5) & 1) * 64;
                float corr = 0.0f;
                for (int t = 0; t < 64; ++t) corr += ep[t] * (float)vcol[t * 32];
                val += corr;
            }
            val /= rowZ[lr];
            const int sAbs = s0 + lr;
            aout[(size_t)sAbs * HIDN + h * 64 + d] = (_Float16)val;
        }
    }
}

extern "C" void kernel_launch(void* const* d_in, const int* in_sizes, int n_in,
                              void* d_out, int out_size, void* d_ws, size_t ws_size,
                              hipStream_t stream) {
    (void)in_sizes; (void)n_in; (void)out_size; (void)ws_size;
    const float* x  = (const float*)d_in[0];
    const float* wq = (const float*)d_in[1];
    const float* bq = (const float*)d_in[2];
    const float* wk = (const float*)d_in[3];
    const float* bk = (const float*)d_in[4];
    const float* wv = (const float*)d_in[5];
    const float* bv = (const float*)d_in[6];
    const float* wo = (const float*)d_in[7];
    const float* bo = (const float*)d_in[8];

    char* ws = (char*)d_ws;
    const size_t MB = 1024 * 1024;
    _Float16* xh   = (_Float16*)(ws + 0 * MB);    // [S][HID]    4 MB
    _Float16* wqh  = (_Float16*)(ws + 4 * MB);    // [HID][HID]  2 MB
    _Float16* wkh  = (_Float16*)(ws + 6 * MB);
    _Float16* wvh  = (_Float16*)(ws + 8 * MB);
    _Float16* woh  = (_Float16*)(ws + 10 * MB);
    _Float16* qh   = (_Float16*)(ws + 12 * MB);   // [H][S][D]   4 MB
    _Float16* kh   = (_Float16*)(ws + 16 * MB);   // [H][S][D]
    _Float16* vT   = (_Float16*)(ws + 20 * MB);   // [H][D][S]   4 MB
    _Float16* aout = (_Float16*)(ws + 24 * MB);   // [S][HID]    4 MB
    float*    vtot = (float*)(ws + 28 * MB);      // [H][D]
    float*    vwin = (float*)(ws + 28 * MB + 65536); // [nW][H][D]

    cvt_f32_f16_kernel<<<2048, 256, 0, stream>>>(x,  xh,  S_LEN * HIDN);
    cvt_f32_f16_kernel<<<1024, 256, 0, stream>>>(wq, wqh, HIDN * HIDN);
    cvt_f32_f16_kernel<<<1024, 256, 0, stream>>>(wk, wkh, HIDN * HIDN);
    cvt_f32_f16_kernel<<<1024, 256, 0, stream>>>(wv, wvh, HIDN * HIDN);
    cvt_f32_f16_kernel<<<1024, 256, 0, stream>>>(wo, woh, HIDN * HIDN);

    dim3 ggrid(HIDN / 128, S_LEN / 128);
    gemm_wmma_kernel<0><<<ggrid, 256, 0, stream>>>(xh, wqh, bq, qh, S_LEN, HIDN, HIDN);
    gemm_wmma_kernel<0><<<ggrid, 256, 0, stream>>>(xh, wkh, bk, kh, S_LEN, HIDN, HIDN);
    gemm_wmma_kernel<1><<<ggrid, 256, 0, stream>>>(xh, wvh, bv, vT, S_LEN, HIDN, HIDN);

    reduce_v_kernel<<<HIDN, 256, 0, stream>>>(vT, vtot, vwin);

    const size_t smem_bytes = (size_t)(64 * LDC + 64 + 64 + 128) * sizeof(float); // ~130 KB (CDNA5 LDS: 320 KB/WGP)
    attn_kernel<<<dim3(8, NH, NWIN), 256, smem_bytes, stream>>>(qh, kh, vT, vtot, vwin, aout);

    gemm_wmma_kernel<2><<<ggrid, 256, 0, stream>>>(aout, woh, bo, d_out, S_LEN, HIDN, HIDN);
}
